// AxialTransformerLayer_50337016709366
// MI455X (gfx1250) — matrix operand order repaired and checked by reading, not compiled, not run on hardware
//
#include <hip/hip_runtime.h>

typedef unsigned short u16;
typedef __attribute__((ext_vector_type(16))) __bf16 v16bf;
typedef __attribute__((ext_vector_type(8)))  float  v8f;
typedef __attribute__((ext_vector_type(4)))  unsigned tdm_g0_t;
typedef __attribute__((ext_vector_type(8)))  int      tdm_g1_t;
typedef __attribute__((ext_vector_type(4)))  int      tdm_g2_t;
typedef __attribute__((ext_vector_type(8)))  int      tdm_g4_t;

#define DEV __device__ __forceinline__

union Frag8  { uint2 u2[4]; v16bf v; };
union Frag16 { u16 s[16];   v16bf v; };

DEV u16 f2bf(float f) {
  unsigned u = __float_as_uint(f);
  u += 0x7FFFu + ((u >> 16) & 1u);     // round-to-nearest-even
  return (u16)(u >> 16);
}

DEV v8f wmma_bf16(v16bf a, v16bf b, v8f c) {
  return __builtin_amdgcn_wmma_f32_16x16x32_bf16(false, a, false, b, (short)0, c, false, false);
}

// The TDM intrinsic writes LDS behind the compiler's back; escape the LDS
// pointer into an asm with a memory clobber so alias analysis treats the
// shared buffer as modified (otherwise the fragment loads get folded away).
DEV void lds_dirty(const u16* p) {
  asm volatile("" : : "v"(p) : "memory");
}

// ---------------------------------------------------------------------------
// Tensor Data Mover: load a 2-D tile [rows x tile0] of bf16 from global to LDS.
// D# built per CDNA5 ISA ch.8: group0 = {count, lds_addr, global_addr, type=2};
// group1 = {data_size=2B, pad cfg, tensor dims, tile dims, dim0 stride}.
// padI/padA are the encoded pad_interval / pad_amount fields:
//   padI: DWORDs between pads (0=2,1=4,2=8,3=16,4=32,...), padA: pad DWORDs-1.
// Issued by one wave; completion tracked with TENSORcnt.
// ---------------------------------------------------------------------------
DEV void tdm_load_2d(unsigned ldsByte, const void* gptr, int strideElems,
                     int rows, int tile0, int padI, int padA) {
  unsigned long long ga = (unsigned long long)gptr;
  tdm_g0_t g0;
  g0[0] = 1u;                                              // count=1, user mode
  g0[1] = ldsByte;                                         // lds_addr
  g0[2] = (unsigned)ga;                                    // global_addr[31:0]
  g0[3] = (unsigned)((ga >> 32) & 0x1FFFFFFull) | 0x80000000u; // addr[56:32]|type=2
  tdm_g1_t g1;
  g1[0] = (int)((1u << 16) | (1u << 20) |
                ((unsigned)padI << 22) | ((unsigned)padA << 25)); // 2B elems, pad en
  const unsigned d0 = (unsigned)strideElems;               // tensor_dim0 (OOB bound)
  const unsigned d1 = (unsigned)rows;                      // tensor_dim1
  g1[1] = (int)((d0 & 0xFFFFu) << 16);                     // dim0[15:0] @ bits63:48
  g1[2] = (int)(((d0 >> 16) & 0xFFFFu) | ((d1 & 0xFFFFu) << 16));
  g1[3] = (int)(((d1 >> 16) & 0xFFFFu) | ((unsigned)tile0 << 16)); // tile_dim0
  g1[4] = rows;                                            // tile_dim1, tile_dim2=0
  g1[5] = strideElems;                                     // tensor_dim0_stride lo32
  g1[6] = 0;
  g1[7] = 0;
  tdm_g2_t z4 = {0, 0, 0, 0};
  tdm_g4_t z8 = {0, 0, 0, 0, 0, 0, 0, 0};
  __builtin_amdgcn_tensor_load_to_lds(g0, g1, z4, z4, z8, 0);
}

// A-matrix fragment (16x32 bf16, MxK): lane L holds row m = base+(L&15);
// K = (L>>4)*16 + 0..15 contiguous.
DEV v16bf lds_frag_a(const u16* tile, int stride, int mBase, int lane) {
  const u16* p = tile + (mBase + (lane & 15)) * stride;
  const int c8 = (lane >> 4) * 16;
  Frag8 f;
  f.u2[0] = *(const uint2*)(p + c8);
  f.u2[1] = *(const uint2*)(p + c8 + 4);
  f.u2[2] = *(const uint2*)(p + c8 + 8);
  f.u2[3] = *(const uint2*)(p + c8 + 12);
  return f.v;
}

// B-matrix fragment (32x16 bf16, KxN): lane L holds column n = base+(L&15),
// K = (L>>4)*16 + 0..15 contiguous.  Tile stored [n][k] row-major (k contiguous).
DEV v16bf lds_frag_b(const u16* tile, int stride, int nBase, int lane) {
  const u16* p = tile + (nBase + (lane & 15)) * stride + (lane >> 4) * 16;
  Frag8 f;
  f.u2[0] = *(const uint2*)(p + 0);
  f.u2[1] = *(const uint2*)(p + 4);
  f.u2[2] = *(const uint2*)(p + 8);
  f.u2[3] = *(const uint2*)(p + 12);
  return f.v;
}

// ---------------------------------------------------------------------------
// Generic bf16 GEMM:  Y[M,N] = act(A[M,K] @ W[N,K]^T + bias) (+resid)
// block = 256 thr (8 waves), tile 128x128, k-step 32. Wave tile 64x32.
// TDM double-buffered tile loads: wave 0 issues DMA for k-step kb+1 while all
// waves run WMMA on k-step kb.
// ---------------------------------------------------------------------------
__global__ __launch_bounds__(256) void k_gemm(
    const u16* __restrict__ A, const u16* __restrict__ W,
    const float* __restrict__ bias, const float* __restrict__ resid,
    float* __restrict__ outF, u16* __restrict__ outB,
    int M, int N, int K, int gelu)
{
  __shared__ u16 smem[4 * 128 * 40];   // A0 | B0 | A1 | B1, 10240 B each
  const int t = threadIdx.x, lane = t & 31, wave = t >> 5;
  const int wi = wave >> 2, wj = wave & 3;           // 2 x 4 wave grid
  const int m0 = blockIdx.y * 128, n0 = blockIdx.x * 128;
  const u16* Abase = A + (size_t)m0 * K;
  const u16* Wbase = W + (size_t)n0 * K;

  v8f acc[4][2] = {};
  const int nk = K >> 5;
  if (wave == 0) {
    tdm_load_2d(0u,      Abase, K, 128, 32, 3, 3);
    tdm_load_2d(10240u,  Wbase, K, 128, 32, 3, 3);
  }
  for (int kbi = 0; kbi < nk; ++kbi) {
    if (wave == 0) __builtin_amdgcn_s_wait_tensorcnt(0);
    __syncthreads();                                   // buffer kbi&1 ready
    lds_dirty(smem);
    if (wave == 0 && kbi + 1 < nk) {
      const unsigned b = ((kbi + 1) & 1) ? 20480u : 0u;
      tdm_load_2d(b,           Abase + (kbi + 1) * 32, K, 128, 32, 3, 3);
      tdm_load_2d(b + 10240u,  Wbase + (kbi + 1) * 32, K, 128, 32, 3, 3);
    }
    const u16* As = smem + ((kbi & 1) ? 10240u : 0u);  // u16 offset = bytes/2
    const u16* Bs = As + 5120;
    v16bf b0 = lds_frag_b(Bs, 40, wj * 32,      lane);
    v16bf b1 = lds_frag_b(Bs, 40, wj * 32 + 16, lane);
#pragma unroll
    for (int ti = 0; ti < 4; ++ti) {
      v16bf af = lds_frag_a(As, 40, wi * 64 + ti * 16, lane);
      acc[ti][0] = wmma_bf16(af, b0, acc[ti][0]);
      acc[ti][1] = wmma_bf16(af, b1, acc[ti][1]);
    }
  }

  const int half = lane >> 4, ln = lane & 15;
#pragma unroll
  for (int ti = 0; ti < 4; ++ti)
#pragma unroll
    for (int tj = 0; tj < 2; ++tj)
#pragma unroll
      for (int rr = 0; rr < 8; ++rr) {
        const int m = m0 + wi * 64 + ti * 16 + rr + 8 * half;
        const int n = n0 + wj * 32 + tj * 16 + ln;
        float v = acc[ti][tj][rr] + bias[n];
        if (gelu) v = 0.5f * v * (1.0f + erff(v * 0.70710678118654752f));
        const size_t idx = (size_t)m * N + n;
        if (resid) v += resid[idx];
        if (outF) outF[idx] = v;
        if (outB) outB[idx] = f2bf(v);
      }
}

// ---------------------------------------------------------------------------
// Row attention scores: aw[h,n,i,j] = scale * sum_{r,d} q[r,i,n,h,d] k[r,j,n,h,d]
// grid (H*N, 256/64); block 256 (8 waves over j), wave tile 64(i) x 32(j).
// Q/K tiles are uniform-stride 2-D tiles (token stride 9216 elems) -> TDM.
// ---------------------------------------------------------------------------
__global__ __launch_bounds__(256) void k_row_scores(
    const u16* __restrict__ qkv, float* __restrict__ aw, float scale)
{
  // Q0(5120B) | K0(20480B) | Q1 | K1
  __shared__ u16 smem[2 * (2560 + 10240)];
  const int t = threadIdx.x, lane = t & 31, wave = t >> 5;
  const int h = blockIdx.x >> 2, n = blockIdx.x & 3;
  const int i0 = blockIdx.y * 64;

  v8f acc[4][2] = {};
  if (wave == 0) {
    tdm_load_2d(0u,     qkv + ((size_t)((0 * 256 + i0) * 4 + n)) * 2304 + h * 64,
                9216, 64, 32, 3, 3);
    tdm_load_2d(5120u,  qkv + ((size_t)(0 * 256 * 4 + n)) * 2304 + 768 + h * 64,
                9216, 256, 32, 3, 3);
  }
  for (int kk = 0; kk < 64; ++kk) {                 // kk = r*2 + ks
    if (wave == 0) __builtin_amdgcn_s_wait_tensorcnt(0);
    __syncthreads();
    lds_dirty(smem);
    if (wave == 0 && kk + 1 < 64) {
      const int r2 = (kk + 1) >> 1, kd2 = ((kk + 1) & 1) * 32;
      const unsigned b = ((kk + 1) & 1) ? 25600u : 0u;
      tdm_load_2d(b, qkv + ((size_t)((r2 * 256 + i0) * 4 + n)) * 2304 + h * 64 + kd2,
                  9216, 64, 32, 3, 3);
      tdm_load_2d(b + 5120u, qkv + ((size_t)(r2 * 256 * 4 + n)) * 2304 + 768 + h * 64 + kd2,
                  9216, 256, 32, 3, 3);
    }
    const u16* Qs = smem + ((kk & 1) ? 12800u : 0u);
    const u16* Ks = Qs + 2560;
    v16bf b0 = lds_frag_b(Ks, 40, wave * 32,      lane);
    v16bf b1 = lds_frag_b(Ks, 40, wave * 32 + 16, lane);
#pragma unroll
    for (int ti = 0; ti < 4; ++ti) {
      v16bf af = lds_frag_a(Qs, 40, ti * 16, lane);
      acc[ti][0] = wmma_bf16(af, b0, acc[ti][0]);
      acc[ti][1] = wmma_bf16(af, b1, acc[ti][1]);
    }
  }
  const int half = lane >> 4, ln = lane & 15;
#pragma unroll
  for (int ti = 0; ti < 4; ++ti)
#pragma unroll
    for (int tj = 0; tj < 2; ++tj)
#pragma unroll
      for (int rr = 0; rr < 8; ++rr) {
        const int i = i0 + ti * 16 + rr + 8 * half;
        const int j = wave * 32 + tj * 16 + ln;
        aw[(((size_t)(h * 4 + n)) * 256 + i) * 256 + j] = acc[ti][tj][rr] * scale;
      }
}

// ---------------------------------------------------------------------------
// Row ctx: ctx[r,i,n,h,d] = sum_j P[h,n,i,j] v[r,j,n,h,d]
// grid (R*H*N, 4); block 256 (waves: 4 over i x 2 over d), wave tile 16(i)x32(d)
// P tile [64x32] stride 256; V tile [32x64] stride 9216, 72-u16 padded rows.
// ---------------------------------------------------------------------------
__global__ __launch_bounds__(256) void k_row_ctx(
    const u16* __restrict__ probs, const u16* __restrict__ qkv, u16* __restrict__ ctx)
{
  // P0(5120B) | V0(4608B) | P1 | V1
  __shared__ u16 smem[2 * (2560 + 2304)];
  const int t = threadIdx.x, lane = t & 31, wave = t >> 5;
  const int bx = blockIdx.x;
  const int r = bx / 48, hn = bx % 48, h = hn >> 2, n = hn & 3;
  const int i0 = blockIdx.y * 64;
  const int wi = wave >> 1, wd = wave & 1;
  const u16* pbase = probs + (((size_t)(h * 4 + n)) * 256 + i0) * 256;

  v8f acc[2] = {};
  if (wave == 0) {
    tdm_load_2d(0u,    pbase, 256, 64, 32, 3, 3);
    tdm_load_2d(5120u, qkv + ((size_t)((r * 256 + 0) * 4 + n)) * 2304 + 1536 + h * 64,
                9216, 32, 64, 4, 3);
  }
  for (int jbi = 0; jbi < 8; ++jbi) {               // jb = jbi*32
    if (wave == 0) __builtin_amdgcn_s_wait_tensorcnt(0);
    __syncthreads();
    lds_dirty(smem);
    if (wave == 0 && jbi + 1 < 8) {
      const int jb2 = (jbi + 1) * 32;
      const unsigned b = ((jbi + 1) & 1) ? 9728u : 0u;
      tdm_load_2d(b,         pbase + jb2, 256, 64, 32, 3, 3);
      tdm_load_2d(b + 5120u, qkv + ((size_t)((r * 256 + jb2) * 4 + n)) * 2304 + 1536 + h * 64,
                  9216, 32, 64, 4, 3);
    }
    const u16* Ps = smem + ((jbi & 1) ? 4864u : 0u);
    const u16* Vs = Ps + 2560;
    v16bf af = lds_frag_a(Ps, 40, wi * 16, lane);
#pragma unroll
    for (int td = 0; td < 2; ++td) {
      Frag16 f;
      const int d  = wd * 32 + td * 16 + (lane & 15);
      const int j0 = (lane >> 4) * 16;
#pragma unroll
      for (int jj = 0; jj < 16; ++jj) f.s[jj] = Vs[(j0 + jj) * 72 + d];
      acc[td] = wmma_bf16(af, f.v, acc[td]);
    }
  }
  const int half = lane >> 4, ln = lane & 15;
#pragma unroll
  for (int td = 0; td < 2; ++td)
#pragma unroll
    for (int rr = 0; rr < 8; ++rr) {
      const int i = i0 + wi * 16 + rr + 8 * half;
      const int d = wd * 32 + td * 16 + ln;
      ctx[((size_t)((r * 256 + i) * 4 + n)) * 768 + h * 64 + d] = f2bf(acc[td][rr]);
    }
}

// ---------------------------------------------------------------------------
// Column attention scores: aw[h,c,n,i,j] = scale * sum_d q[i,c,n,h,d] k[j,c,n,h,d]
// grid C*N, block 128 (4 waves); each wave: heads {w, w+4, w+8}, 32x32 out, K=64.
// Fragments gathered directly from global memory.
// ---------------------------------------------------------------------------
__global__ __launch_bounds__(128) void k_col_scores(
    const u16* __restrict__ qkv, float* __restrict__ aw, float scale)
{
  const int t = threadIdx.x, lane = t & 31, wave = t >> 5;
  const int c = blockIdx.x >> 2, n = blockIdx.x & 3;

  for (int h = wave; h < 12; h += 4) {
    v8f acc[2][2] = {};
#pragma unroll
    for (int ks = 0; ks < 2; ++ks) {
      v16bf aq[2], bk[2];
#pragma unroll
      for (int ti = 0; ti < 2; ++ti) {
        const int i  = ti * 16 + (lane & 15);
        const int c8 = (lane >> 4) * 16;
        const u16* p = qkv + ((size_t)((i * 256 + c) * 4 + n)) * 2304 + h * 64 + ks * 32;
        Frag8 f;
        f.u2[0] = *(const uint2*)(p + c8);      f.u2[1] = *(const uint2*)(p + c8 + 4);
        f.u2[2] = *(const uint2*)(p + c8 + 8);  f.u2[3] = *(const uint2*)(p + c8 + 12);
        aq[ti] = f.v;
      }
#pragma unroll
      for (int tj = 0; tj < 2; ++tj) {
        const int j = tj * 16 + (lane & 15);
        const u16* p = qkv + ((size_t)((j * 256 + c) * 4 + n)) * 2304 + 768 + h * 64 +
                       ks * 32 + (lane >> 4) * 16;
        Frag8 f;
        f.u2[0] = *(const uint2*)(p + 0); f.u2[1] = *(const uint2*)(p + 4);
        f.u2[2] = *(const uint2*)(p + 8); f.u2[3] = *(const uint2*)(p + 12);
        bk[tj] = f.v;
      }
#pragma unroll
      for (int ti = 0; ti < 2; ++ti)
#pragma unroll
        for (int tj = 0; tj < 2; ++tj)
          acc[ti][tj] = wmma_bf16(aq[ti], bk[tj], acc[ti][tj]);
    }
    const int half = lane >> 4, ln = lane & 15;
    float* base = aw + ((size_t)((h * 256 + c) * 4 + n)) * 1024;
#pragma unroll
    for (int ti = 0; ti < 2; ++ti)
#pragma unroll
      for (int tj = 0; tj < 2; ++tj)
#pragma unroll
        for (int rr = 0; rr < 8; ++rr) {
          const int i = ti * 16 + rr + 8 * half;
          const int j = tj * 16 + ln;
          base[i * 32 + j] = acc[ti][tj][rr] * scale;
        }
  }
}

// ---------------------------------------------------------------------------
// Column ctx: ctx[i,c,n,h,d] = sum_j P[h,c,n,i,j] v[j,c,n,h,d]; one wave per (h,c,n)
// ---------------------------------------------------------------------------
__global__ __launch_bounds__(256) void k_col_ctx(
    const u16* __restrict__ probs, const u16* __restrict__ qkv, u16* __restrict__ ctx)
{
  const int t = threadIdx.x, lane = t & 31, wave = t >> 5;
  const int g = blockIdx.x * 8 + wave;            // [0, 12288)
  const int h = g >> 10, cn = g & 1023, c = cn >> 2, n = cn & 3;
  const u16* pbase = probs + (size_t)g * 1024;

  v8f acc[2][4] = {};
  v16bf aq[2];
#pragma unroll
  for (int ti = 0; ti < 2; ++ti) {
    const int i  = ti * 16 + (lane & 15);
    const int c8 = (lane >> 4) * 16;
    const u16* p = pbase + i * 32;
    Frag8 f;
    f.u2[0] = *(const uint2*)(p + c8);      f.u2[1] = *(const uint2*)(p + c8 + 4);
    f.u2[2] = *(const uint2*)(p + c8 + 8);  f.u2[3] = *(const uint2*)(p + c8 + 12);
    aq[ti] = f.v;
  }
#pragma unroll
  for (int td = 0; td < 4; ++td) {
    Frag16 f;
    const int d  = td * 16 + (lane & 15);
    const int j0 = (lane >> 4) * 16;
#pragma unroll
    for (int jj = 0; jj < 16; ++jj)
      f.s[jj] = qkv[((size_t)(((j0 + jj) * 256 + c) * 4 + n)) * 2304 + 1536 + h * 64 + d];
#pragma unroll
    for (int ti = 0; ti < 2; ++ti)
      acc[ti][td] = wmma_bf16(aq[ti], f.v, acc[ti][td]);
  }
  const int half = lane >> 4, ln = lane & 15;
#pragma unroll
  for (int ti = 0; ti < 2; ++ti)
#pragma unroll
    for (int td = 0; td < 4; ++td)
#pragma unroll
      for (int rr = 0; rr < 8; ++rr) {
        const int i = ti * 16 + rr + 8 * half;
        const int d = td * 16 + ln;
        ctx[((size_t)((i * 256 + c) * 4 + n)) * 768 + h * 64 + d] = f2bf(acc[ti][td][rr]);
      }
}

// ---------------------------------------------------------------------------
// Softmax over rows of 256 (one wave/row) and 32 (one wave/row), bf16 out.
// ---------------------------------------------------------------------------
__global__ __launch_bounds__(256) void k_softmax256(const float* __restrict__ s, u16* __restrict__ p)
{
  const int row = blockIdx.x * 8 + (threadIdx.x >> 5);
  const int lane = threadIdx.x & 31;
  const float* sp = s + (size_t)row * 256;
  float v[8], mx = -3.4e38f;
#pragma unroll
  for (int k = 0; k < 8; ++k) { v[k] = sp[lane + 32 * k]; mx = fmaxf(mx, v[k]); }
#pragma unroll
  for (int off = 16; off >= 1; off >>= 1) mx = fmaxf(mx, __shfl_xor(mx, off, 32));
  float sum = 0.f;
#pragma unroll
  for (int k = 0; k < 8; ++k) { v[k] = __expf(v[k] - mx); sum += v[k]; }
#pragma unroll
  for (int off = 16; off >= 1; off >>= 1) sum += __shfl_xor(sum, off, 32);
  const float rinv = 1.0f / sum;
  u16* pp = p + (size_t)row * 256;
#pragma unroll
  for (int k = 0; k < 8; ++k) pp[lane + 32 * k] = f2bf(v[k] * rinv);
}

__global__ __launch_bounds__(256) void k_softmax32(const float* __restrict__ s, u16* __restrict__ p)
{
  const int row = blockIdx.x * 8 + (threadIdx.x >> 5);
  const int lane = threadIdx.x & 31;
  float v = s[(size_t)row * 32 + lane];
  float mx = v;
#pragma unroll
  for (int off = 16; off >= 1; off >>= 1) mx = fmaxf(mx, __shfl_xor(mx, off, 32));
  v = __expf(v - mx);
  float sum = v;
#pragma unroll
  for (int off = 16; off >= 1; off >>= 1) sum += __shfl_xor(sum, off, 32);
  p[(size_t)row * 32 + lane] = f2bf(v / sum);
}

// ---------------------------------------------------------------------------
// LayerNorm over E=768, bf16 out.  One block per token.
// ---------------------------------------------------------------------------
__global__ __launch_bounds__(256) void k_ln(const float* __restrict__ x,
    const float* __restrict__ w, const float* __restrict__ b, u16* __restrict__ out)
{
  __shared__ float rs[256], rs2[256];
  const int t = threadIdx.x;
  const float* xp = x + (size_t)blockIdx.x * 768;
  float s = 0.f, s2 = 0.f;
#pragma unroll
  for (int k = 0; k < 3; ++k) { const float v = xp[t + 256 * k]; s += v; s2 += v * v; }
  rs[t] = s; rs2[t] = s2; __syncthreads();
  for (int off = 128; off > 0; off >>= 1) {
    if (t < off) { rs[t] += rs[t + off]; rs2[t] += rs2[t + off]; }
    __syncthreads();
  }
  const float mean = rs[0] * (1.f / 768.f);
  const float var  = rs2[0] * (1.f / 768.f) - mean * mean;
  const float inv  = rsqrtf(var + 1e-12f);
  u16* op = out + (size_t)blockIdx.x * 768;
#pragma unroll
  for (int k = 0; k < 3; ++k) { const int i = t + 256 * k; op[i] = f2bf((xp[i] - mean) * inv * w[i] + b[i]); }
}

__global__ void k_cvt(const float* __restrict__ in, u16* __restrict__ out, int n)
{
  const int i = blockIdx.x * 256 + threadIdx.x;
  if (i < n) out[i] = f2bf(in[i]);
}

// ---------------------------------------------------------------------------

extern "C" void kernel_launch(void* const* d_in, const int* in_sizes, int n_in,
                              void* d_out, int out_size, void* d_ws, size_t ws_size,
                              hipStream_t stream)
{
  (void)in_sizes; (void)n_in; (void)out_size; (void)ws_size;
  const float* x     = (const float*)d_in[0];
  const float* ln_w  = (const float*)d_in[1];
  const float* ln_b  = (const float*)d_in[2];
  const float* row_w = (const float*)d_in[3];
  const float* row_b = (const float*)d_in[4];
  const float* col_w = (const float*)d_in[5];
  const float* col_b = (const float*)d_in[6];
  const float* fc1_w = (const float*)d_in[7];
  const float* fc1_b = (const float*)d_in[8];
  const float* fc2_w = (const float*)d_in[9];
  const float* fc2_b = (const float*)d_in[10];
  float* out = (float*)d_out;
  char*  ws  = (char*)d_ws;

  size_t off = 0;
  auto take = [&](size_t bytes) -> char* {
    char* p = ws + off; off += (bytes + 255) & ~(size_t)255; return p;
  };
  u16* wRow = (u16*)take((size_t)4 * 768 * 768 * 2);
  u16* wCol = (u16*)take((size_t)4 * 768 * 768 * 2);
  u16* wFc1 = (u16*)take((size_t)3072 * 768 * 2);
  u16* wFc2 = (u16*)take((size_t)3072 * 768 * 2);
  u16* hbuf = (u16*)take((size_t)32768 * 768 * 2);
  char* big0 = take((size_t)32768 * 2304 * 2);      // qkv   (151 MB)
  char* big1 = take((size_t)12 * 1024 * 1024 * 4);  // scores (50 MB, col max)
  char* big2 = take((size_t)12 * 1024 * 1024 * 2);  // probs  (25 MB, col max)
  u16* ctxb = (u16*)take((size_t)32768 * 768 * 2);
  u16*   qkv    = (u16*)big0;
  float* scores = (float*)big1;
  u16*   probs  = (u16*)big2;
  u16*   h1     = (u16*)big0;  // FFN hidden (201 MB) overlays dead qkv+scores+probs (226 MB)

  // weight conversion fp32 -> bf16
  k_cvt<<<(4 * 768 * 768 + 255) / 256, 256, 0, stream>>>(row_w, wRow, 4 * 768 * 768);
  k_cvt<<<(4 * 768 * 768 + 255) / 256, 256, 0, stream>>>(col_w, wCol, 4 * 768 * 768);
  k_cvt<<<(3072 * 768 + 255) / 256, 256, 0, stream>>>(fc1_w, wFc1, 3072 * 768);
  k_cvt<<<(3072 * 768 + 255) / 256, 256, 0, stream>>>(fc2_w, wFc2, 3072 * 768);

  const float rowScale = 0.02209708691207961f;  // D^-0.5 / sqrt(R) = 1/(8*sqrt(32))
  const float colScale = 0.125f;                // D^-0.5

  // ---- Row attention block ----
  k_ln<<<32768, 256, 0, stream>>>(x, ln_w, ln_b, hbuf);
  k_gemm<<<dim3(18, 256), 256, 0, stream>>>(hbuf, wRow, row_b, nullptr,
                                            nullptr, qkv, 32768, 2304, 768, 0);
  k_row_scores<<<dim3(48, 4), 256, 0, stream>>>(qkv, scores, rowScale);
  k_softmax256<<<12288 / 8, 256, 0, stream>>>(scores, probs);
  k_row_ctx<<<dim3(1536, 4), 256, 0, stream>>>(probs, qkv, ctxb);
  k_gemm<<<dim3(6, 256), 256, 0, stream>>>(ctxb, wRow + (size_t)3 * 768 * 768, row_b + 3 * 768,
                                           x, out, nullptr, 32768, 768, 768, 0);

  // ---- Column attention block ----
  k_ln<<<32768, 256, 0, stream>>>(out, ln_w + 768, ln_b + 768, hbuf);
  k_gemm<<<dim3(18, 256), 256, 0, stream>>>(hbuf, wCol, col_b, nullptr,
                                            nullptr, qkv, 32768, 2304, 768, 0);
  k_col_scores<<<1024, 128, 0, stream>>>(qkv, scores, colScale);
  k_softmax32<<<393216 / 8, 256, 0, stream>>>(scores, probs);
  k_col_ctx<<<1536, 256, 0, stream>>>(probs, qkv, ctxb);
  k_gemm<<<dim3(6, 256), 256, 0, stream>>>(ctxb, wCol + (size_t)3 * 768 * 768, col_b + 3 * 768,
                                           out, out, nullptr, 32768, 768, 768, 0);

  // ---- FFN block ----
  k_ln<<<32768, 256, 0, stream>>>(out, ln_w + 2 * 768, ln_b + 2 * 768, hbuf);
  k_gemm<<<dim3(24, 256), 256, 0, stream>>>(hbuf, wFc1, fc1_b, nullptr,
                                            nullptr, h1, 32768, 3072, 768, 1);
  k_gemm<<<dim3(6, 256), 256, 0, stream>>>(h1, wFc2, fc2_b, out,
                                           out, nullptr, 32768, 768, 3072, 0);
}
